// HyperEdgeAttention_37177236914496
// MI455X (gfx1250) — matrix operand order repaired and checked by reading, not compiled, not run on hardware
//
#include <hip/hip_runtime.h>
#include <hip/hip_bf16.h>

// ---------------------------------------------------------------------------
// HyperEdgeAttention for MI455X (gfx1250, wave32, WMMA)
//
// Pipeline:
//   k_cvt      : h  (fp32 [N,128])  -> hb (bf16)
//   k_pack     : Wq/Wk/Wv/Wo -> bf16 packs in the WMMA B-matrix VGPR layout
//   k_init     : m=-inf, denom=0, hmid=0
//   k_gemm4    : QKV = h @ [Wq;Wk;Wv]^T + b  (16x64 strip/wave, 16x WMMA)
//   k_score    : score[e,h] = Q[src].K[dst]*scale, atomic segment-max
//   k_expnorm  : ex = exp(score - m[dst]), atomic segment-sum (denom)
//   k_scatter  : hmid[dst] += V[src] * ex/denom  (global_atomic_add_f32)
//   k_cvt      : hmid -> bf16
//   k_gemm4    : out = hmid @ Wo^T + bo
// ---------------------------------------------------------------------------

#define NNODES 50000
#define NEDGES 1600000
#define HEADS  8
#define HDIM   16
#define SCALE  0.25f            // 16^-0.5

typedef __attribute__((ext_vector_type(16))) __bf16          v16bf;
typedef __attribute__((ext_vector_type(8)))  float           v8f;
typedef __attribute__((ext_vector_type(8)))  unsigned short  v8us;

union ABV { v16bf bv; v8us us[2]; };

// fp32 -> bf16, round-to-nearest-even, in the integer domain
__device__ __forceinline__ unsigned short f2bf(float f) {
  unsigned u = __float_as_uint(f);
  unsigned r = u + 0x7FFFu + ((u >> 16) & 1u);
  return (unsigned short)(r >> 16);
}

// ---------------------------------------------------------------------------
// Workspace layout (bytes, all 256B aligned)
// ---------------------------------------------------------------------------
#define OFF_HB      0u            // bf16 h          [N,128]        12,800,000
#define OFF_WPQKV   12800000u     // bf16 Wqkv pack  24*4*32*16         98,304
#define OFF_WPO     12898304u     // bf16 Wo pack     8*4*32*16         32,768
#define OFF_BIAS    12931072u     // f32 biascat [384 qkv | 128 o]       2,048
#define OFF_QKV     12933120u     // f32 QKV         [N,384]        76,800,000
#define OFF_SCORE   89733120u     // f32 score/ex    [E,8]          51,200,000
#define OFF_M       140933120u    // f32 seg max     [N,8]           1,600,000
#define OFF_DEN     142533120u    // f32 seg sum     [N,8]           1,600,000
#define OFF_HMID    144133120u    // f32 attn out    [N,128]        25,600,000
#define OFF_HMIDB   169733120u    // bf16 attn out   [N,128]        12,800,000

// ---------------------------------------------------------------------------
__global__ void __launch_bounds__(256)
k_cvt(const float* __restrict__ x, unsigned short* __restrict__ y, int n) {
  for (int i = blockIdx.x * 256 + threadIdx.x; i < n; i += gridDim.x * 256)
    y[i] = f2bf(x[i]);
}

// Pack weights into B-matrix (32x16, bf16) VGPR layout:
//   entry index = ((nt*4 + ks)*32 + lane)*16 + e
//   N column    = nt*16 + (lane&15)
//   K row       = ks*32 + (lane<16 ? 0 : 16) + e
__global__ void __launch_bounds__(256)
k_pack(const float* __restrict__ Wq, const float* __restrict__ Wk,
       const float* __restrict__ Wv, const float* __restrict__ Wo,
       const float* __restrict__ bq, const float* __restrict__ bk,
       const float* __restrict__ bv, const float* __restrict__ bo,
       unsigned short* __restrict__ wpqkv, unsigned short* __restrict__ wpo,
       float* __restrict__ biascat) {
  int tid = blockIdx.x * 256 + threadIdx.x;
  if (tid < 49152) {                       // 24 tiles of Wcat = [Wq;Wk;Wv]
    int e = tid & 15, lane = (tid >> 4) & 31, ks = (tid >> 9) & 3, nt = tid >> 11;
    int nrow = nt * 16 + (lane & 15);
    int K = ks * 32 + ((lane < 16) ? 0 : 16) + e;
    float w = (nrow < 128) ? Wq[nrow * 128 + K]
            : (nrow < 256) ? Wk[(nrow - 128) * 128 + K]
                           : Wv[(nrow - 256) * 128 + K];
    wpqkv[tid] = f2bf(w);
  } else if (tid < 65536) {                // 8 tiles of Wo
    int t2 = tid - 49152;
    int e = t2 & 15, lane = (t2 >> 4) & 31, ks = (t2 >> 9) & 3, nt = t2 >> 11;
    int nrow = nt * 16 + (lane & 15);
    int K = ks * 32 + ((lane < 16) ? 0 : 16) + e;
    wpo[t2] = f2bf(Wo[nrow * 128 + K]);
  } else if (tid < 65536 + 512) {          // bias concat
    int c = tid - 65536;
    float v = (c < 128) ? bq[c] : (c < 256) ? bk[c - 128]
            : (c < 384) ? bv[c - 256] : bo[c - 384];
    biascat[c] = v;
  }
}

__global__ void __launch_bounds__(256)
k_init(float* __restrict__ mbuf, float* __restrict__ den,
       float* __restrict__ hmid, int nmh, int nhid) {
  for (int i = blockIdx.x * 256 + threadIdx.x; i < nhid; i += gridDim.x * 256) {
    hmid[i] = 0.0f;
    if (i < nmh) { mbuf[i] = -__builtin_inff(); den[i] = 0.0f; }
  }
}

// ---------------------------------------------------------------------------
// bf16 WMMA GEMM, 16x64 output strip per wave (4 N-tiles share one A frag).
// K=128 as 4 steps; 16 v_wmma_f32_16x16x32_bf16 per wave.
// A layout (16-bit A 16x32, same k order as the k_pack B): lane<16 row=lane, K {0..15};
// lane>=16 row=lane-16, K {16..31}  -> two 16B loads per lane per step.
// EXEC must be all-ones for WMMA: tail guard is wave-uniform, and the M grid
// is exact (50000 = 3125*16).
// ---------------------------------------------------------------------------
__global__ void __launch_bounds__(256)
k_gemm4(const unsigned short* __restrict__ Abf,   // bf16 [rows,128] row-major
        const unsigned short* __restrict__ Bpack, // packed by k_pack
        const float* __restrict__ bias,           // [ntiles*16]
        float* __restrict__ Cout,                 // f32 [rows, ldc]
        int ntiles, int ldc, int mtiles) {
  int lane = threadIdx.x & 31;
  int wave = threadIdx.x >> 5;
  int cg = ntiles >> 2;                          // column groups of 4 tiles
  int t = blockIdx.x * 8 + wave;
  if (t >= mtiles * cg) return;                  // wave-uniform
  int mt = t / cg;
  int ng = t % cg;

  const unsigned short* arow = Abf + (mt * 16 + (lane & 15)) * 128;
  int kA0 = (lane < 16) ? 0 : 16;

  v8f acc[4] = {};
#pragma unroll
  for (int ks = 0; ks < 4; ++ks) {
    ABV a;
    a.us[0] = *(const v8us*)(arow + ks * 32 + kA0);
    a.us[1] = *(const v8us*)(arow + ks * 32 + kA0 + 8);
    // tile (ng*4+j) pack base; per-tile stride = 4*32*16 = 2048 ushorts
    const unsigned short* bp = Bpack + ((((ng * 4) * 4 + ks) * 32 + lane) << 4);
#pragma unroll
    for (int j = 0; j < 4; ++j) {
      ABV b;
      b.us[0] = *(const v8us*)(bp + j * 2048);
      b.us[1] = *(const v8us*)(bp + j * 2048 + 8);
      acc[j] = __builtin_amdgcn_wmma_f32_16x16x32_bf16(
          /*neg_a=*/false, a.bv, /*neg_b=*/false, b.bv,
          /*c_mod=*/(short)0, acc[j], /*reuse_a=*/false, /*reuse_b=*/false);
    }
  }

  // C/D layout: VGPR r, lanes 0-15 -> M=r, lanes 16-31 -> M=r+8, N=lane&15
  int rbase = mt * 16 + ((lane < 16) ? 0 : 8);
#pragma unroll
  for (int j = 0; j < 4; ++j) {
    int col = (ng * 4 + j) * 16 + (lane & 15);
    float bb = bias[col];
#pragma unroll
    for (int r = 0; r < 8; ++r)
      Cout[(rbase + r) * ldc + col] = acc[j][r] + bb;
  }
}

// ---------------------------------------------------------------------------
// Edge phase. Q/K/V (77MB) live in the 192MB L2, so per-edge gathers stay
// on-chip. One thread per (edge, head); 8 consecutive threads share one edge
// so their Q/K float4 loads cover one contiguous 512B node row.
// ---------------------------------------------------------------------------
__device__ __forceinline__ void atomicMaxFloat(float* addr, float val) {
  if (val >= 0.0f) atomicMax((int*)addr, __float_as_int(val));
  else             atomicMin((unsigned int*)addr, __float_as_uint(val));
}

__global__ void __launch_bounds__(256)
k_score(const int* __restrict__ src, const int* __restrict__ dst,
        const float* __restrict__ qkv, float* __restrict__ score,
        float* __restrict__ mbuf, int e8) {
  int tid = blockIdx.x * 256 + threadIdx.x;
  if (tid >= e8) return;
  int e = tid >> 3, hh = tid & 7;
  int s = src[e], d = dst[e];
  const float4* qp = (const float4*)(qkv + s * 384 + hh * 16);
  const float4* kp = (const float4*)(qkv + d * 384 + 128 + hh * 16);
  float sc = 0.0f;
#pragma unroll
  for (int i = 0; i < 4; ++i) {
    float4 q = qp[i], k = kp[i];
    sc += q.x * k.x + q.y * k.y + q.z * k.z + q.w * k.w;
  }
  sc *= SCALE;
  score[tid] = sc;
  atomicMaxFloat(mbuf + d * 8 + hh, sc);
}

__global__ void __launch_bounds__(256)
k_expnorm(const int* __restrict__ dst, float* __restrict__ score,
          const float* __restrict__ mbuf, float* __restrict__ den, int e8) {
  int tid = blockIdx.x * 256 + threadIdx.x;
  if (tid >= e8) return;
  int e = tid >> 3, hh = tid & 7;
  int d = dst[e];
  float ex = __expf(score[tid] - mbuf[d * 8 + hh]);
  score[tid] = ex;
  atomicAdd(den + d * 8 + hh, ex);
}

__global__ void __launch_bounds__(256)
k_scatter(const int* __restrict__ src, const int* __restrict__ dst,
          const float* __restrict__ score, const float* __restrict__ den,
          const float* __restrict__ qkv, float* __restrict__ hmid, int e8) {
  int tid = blockIdx.x * 256 + threadIdx.x;
  if (tid >= e8) return;
  int e = tid >> 3, hh = tid & 7;
  int s = src[e], d = dst[e];
  float a = score[tid] / den[d * 8 + hh];
  const float4* vp = (const float4*)(qkv + s * 384 + 256 + hh * 16);
  float* op = hmid + d * 128 + hh * 16;
#pragma unroll
  for (int i = 0; i < 4; ++i) {
    float4 v = vp[i];
    atomicAdd(op + 4 * i + 0, v.x * a);
    atomicAdd(op + 4 * i + 1, v.y * a);
    atomicAdd(op + 4 * i + 2, v.z * a);
    atomicAdd(op + 4 * i + 3, v.w * a);
  }
}

// ---------------------------------------------------------------------------
extern "C" void kernel_launch(void* const* d_in, const int* in_sizes, int n_in,
                              void* d_out, int out_size, void* d_ws, size_t ws_size,
                              hipStream_t stream) {
  const float* h  = (const float*)d_in[0];
  const int* src  = (const int*)d_in[1];
  const int* dst  = (const int*)d_in[2];
  const float* Wq = (const float*)d_in[3];
  const float* bq = (const float*)d_in[4];
  const float* Wk = (const float*)d_in[5];
  const float* bk = (const float*)d_in[6];
  const float* Wv = (const float*)d_in[7];
  const float* bv = (const float*)d_in[8];
  const float* Wo = (const float*)d_in[9];
  const float* bo = (const float*)d_in[10];
  float* out = (float*)d_out;

  char* ws = (char*)d_ws;
  unsigned short* hb    = (unsigned short*)(ws + OFF_HB);
  unsigned short* wpqkv = (unsigned short*)(ws + OFF_WPQKV);
  unsigned short* wpo   = (unsigned short*)(ws + OFF_WPO);
  float* biascat        = (float*)(ws + OFF_BIAS);
  float* qkv            = (float*)(ws + OFF_QKV);
  float* score          = (float*)(ws + OFF_SCORE);
  float* mbuf           = (float*)(ws + OFF_M);
  float* den            = (float*)(ws + OFF_DEN);
  float* hmid           = (float*)(ws + OFF_HMID);
  unsigned short* hmidb = (unsigned short*)(ws + OFF_HMIDB);

  const int e8 = NEDGES * HEADS;                 // 12.8M
  const int edgeBlocks = (e8 + 255) / 256;       // 50000

  // h -> bf16
  k_cvt<<<(NNODES * 128 + 255) / 256, 256, 0, stream>>>(h, hb, NNODES * 128);
  // weight packing + bias concat
  k_pack<<<(66048 + 255) / 256, 256, 0, stream>>>(Wq, Wk, Wv, Wo, bq, bk, bv, bo,
                                                  wpqkv, wpo, biascat);
  // init segment buffers + accumulator
  k_init<<<(NNODES * 128 + 255) / 256, 256, 0, stream>>>(mbuf, den, hmid,
                                                         NNODES * 8, NNODES * 128);
  // QKV = h @ [Wq;Wk;Wv]^T + b : 3125 rows x 6 column-groups = 18750 waves
  k_gemm4<<<(18750 + 7) / 8, 256, 0, stream>>>(hb, wpqkv, biascat, qkv,
                                               24, 384, 3125);
  // edge attention
  k_score  <<<edgeBlocks, 256, 0, stream>>>(src, dst, qkv, score, mbuf, e8);
  k_expnorm<<<edgeBlocks, 256, 0, stream>>>(dst, score, mbuf, den, e8);
  k_scatter<<<edgeBlocks, 256, 0, stream>>>(src, dst, score, den, qkv, hmid, e8);
  // hmid -> bf16
  k_cvt<<<(NNODES * 128 + 255) / 256, 256, 0, stream>>>(hmid, hmidb, NNODES * 128);
  // out = hmid @ Wo^T + bo : 3125 rows x 2 column-groups = 6250 waves
  k_gemm4<<<(6250 + 7) / 8, 256, 0, stream>>>(hmidb, wpo, biascat + 384, out,
                                              8, 128, 3125);
}
